// DMoNClustering_41755672051945
// MI455X (gfx1250) — matrix operand order repaired and checked by reading, not compiled, not run on hardware
//
#include <hip/hip_runtime.h>
#include <hip/hip_bf16.h>

// ---------------- problem constants ----------------
constexpr int NN = 50000;   // nodes
constexpr int EE = 800000;  // edges
constexpr int DD = 512;     // embedding dim
constexpr int HH = 256;     // hidden dim
constexpr int KK = 16;      // clusters

// ---------------- workspace layout (float units for f32 region) ----------------
constexpr size_t NPAD   = 50048;                     // N rounded to 64-float alignment
constexpr size_t SCAL_O = 0;                         // 64 floats of scalars:
                                                     // [0..15]=csize [16..31]=dS [32]=ent [33]=msum [34]=trace
constexpr size_t DEGD_O = 64;                        // deg by dst (N)
constexpr size_t DEGS_O = DEGD_O + NPAD;             // deg by src (N)
constexpr size_t DIS_O  = DEGS_O + NPAD;             // rsqrt(1+deg_dst) (N)
constexpr size_t B1_O   = DIS_O + NPAD;              // f32 [N,H]
constexpr size_t B2_O   = B1_O + (size_t)NN * HH;    // f32 [N,H]
constexpr size_t B3_O   = B2_O + (size_t)NN * HH;    // f32 [N,H]
constexpr size_t F32_END = B3_O + (size_t)NN * HH;
// f16 region (offsets in _Float16 units, based at wsf + F32_END)
constexpr size_t XH_O  = 0;                          // f16 [N,D]
constexpr size_t HF_O  = XH_O + (size_t)NN * DD;     // f16 [N,H] (reused)
constexpr size_t W1T_O = HF_O + (size_t)NN * HH;     // f16 [H,D]  (= W1^T)
constexpr size_t W2T_O = W1T_O + (size_t)HH * DD;    // f16 [H,H]  (= W2^T)
constexpr size_t WAT_O = W2T_O + (size_t)HH * HH;    // f16 [K,H]  (= Wa^T)

// ---------------- vector types ----------------
typedef __attribute__((ext_vector_type(16))) _Float16 v16h;
typedef __attribute__((ext_vector_type(8)))  _Float16 v8h;
typedef __attribute__((ext_vector_type(8)))  float    v8f;
typedef __attribute__((ext_vector_type(4)))  unsigned int u32x4;
typedef __attribute__((ext_vector_type(4)))  int      i32x4;
typedef __attribute__((ext_vector_type(8)))  int      i32x8;

__device__ __forceinline__ float selu_f(float x) {
  constexpr float scale = 1.0507009873554805f;
  constexpr float alpha = 1.6732632423543772f;
  return x > 0.f ? scale * x : scale * alpha * (__expf(x) - 1.f);
}

// ---------------- utility kernels ----------------
__global__ void k_zero(float* __restrict__ p, size_t n) {
  size_t i = (size_t)blockIdx.x * blockDim.x + threadIdx.x;
  if (i < n) p[i] = 0.f;
}

__global__ void k_deg(const int* __restrict__ src, const int* __restrict__ dst,
                      float* __restrict__ deg_dst, float* __restrict__ deg_src) {
  int e = blockIdx.x * blockDim.x + threadIdx.x;
  if (e >= EE) return;
  atomicAdd(&deg_dst[dst[e]], 1.f);
  atomicAdd(&deg_src[src[e]], 1.f);
}

__global__ void k_dis(const float* __restrict__ deg_dst, float* __restrict__ dis) {
  int n = blockIdx.x * blockDim.x + threadIdx.x;
  if (n < NN) dis[n] = rsqrtf(1.f + deg_dst[n]);
}

__global__ void k_cvt_f16(const float* __restrict__ src, _Float16* __restrict__ dst, size_t n) {
  size_t i = (size_t)blockIdx.x * blockDim.x + threadIdx.x;
  if (i < n) dst[i] = (_Float16)src[i];
}

// dst[c * rows + r] = src[r * cols + c]   (produces [cols, rows] f16 from [rows, cols] f32)
__global__ void k_transpose(const float* __restrict__ src, _Float16* __restrict__ dst,
                            int rows, int cols) {
  int i = blockIdx.x * blockDim.x + threadIdx.x;
  if (i >= rows * cols) return;
  int c = i / rows, r = i % rows;
  dst[i] = (_Float16)src[(size_t)r * cols + c];
}

// ---------------- TDM stage: contiguous 1-D tile (nbytes at gaddr) -> LDS ----------------
__device__ __forceinline__ void tdm_stage_1d(const _Float16* gsrc, _Float16* lds_dst,
                                             unsigned int nelem /* f16 elements */) {
#if __has_builtin(__builtin_amdgcn_tensor_load_to_lds)
  unsigned long long ga = (unsigned long long)(const void*)gsrc;
  unsigned int lds = (unsigned int)(size_t)(void*)lds_dst;  // low 32 bits = LDS offset
  u32x4 g0;
  g0[0] = 1u;                                               // count=1 (valid), user mode
  g0[1] = lds;                                              // lds_addr
  g0[2] = (unsigned int)(ga & 0xffffffffull);               // global_addr[31:0]
  g0[3] = (unsigned int)((ga >> 32) & 0x01ffffffull) | (2u << 30);  // addr[56:32] | type=2
  i32x8 g1;
  g1[0] = (int)(1u << 16);                                  // data_size=1 (2 bytes)
  g1[1] = (int)((nelem & 0xffffu) << 16);                   // tensor_dim0[15:0]
  g1[2] = (int)((nelem >> 16) | (1u << 16));                // tensor_dim0[31:16] | tensor_dim1=1
  g1[3] = (int)(nelem << 16);                               // tile_dim0 (1-D tile)
  g1[4] = 0;                                                // tile_dim1=0, tile_dim2=0
  g1[5] = (int)nelem;                                       // tensor_dim0_stride lo
  g1[6] = 0;
  g1[7] = 0;
  i32x4 g2 = {};
  i32x4 g3 = {};
#if __clang_major__ >= 23
  i32x8 g4 = {};
  __builtin_amdgcn_tensor_load_to_lds(g0, g1, g2, g3, g4, 0);
#else
  __builtin_amdgcn_tensor_load_to_lds(g0, g1, g2, g3, 0);
#endif
  __builtin_amdgcn_s_wait_tensorcnt(0);
#else
  (void)gsrc; (void)lds_dst; (void)nelem;
#endif
}

// ---------------- WMMA GEMM with LDS-staged A block ----------------
// C[M,Nc] = A[M,K] * B^T[Nc,K].  One workgroup = 16 waves = one 16-row block of A
// (staged once into LDS via the Tensor Data Mover) x all Ntiles column tiles.
// A-fragment (16-bit 16x32): lane l: m=l%16, g=l/16; e=0..15 -> K=k0+16g+e
// B-fragment (16-bit 32x16): lane l: n=l%16, g=l/16; e=0..15 -> K=k0+16g+e
template <int KDIM>
__global__ void k_gemm_wmma_lds(const _Float16* __restrict__ A, const _Float16* __restrict__ Bt,
                                float* __restrict__ C, int Ntiles) {
  __shared__ _Float16 As[16 * KDIM];
  const int mt   = blockIdx.x;
  const int tid  = threadIdx.x;
  const int lane = tid & 31;
  const int w    = tid >> 5;   // wave id == column tile nt

#if __has_builtin(__builtin_amdgcn_tensor_load_to_lds)
  if (tid < 32) {
    tdm_stage_1d(A + (size_t)mt * 16 * KDIM, As, 16u * (unsigned)KDIM);
  }
#else
  {
    const v8h* __restrict__ src = (const v8h*)(A + (size_t)mt * 16 * KDIM);
    v8h* dst = (v8h*)As;
    for (int i = tid; i < (16 * KDIM) / 8; i += blockDim.x) dst[i] = src[i];
  }
#endif
  __syncthreads();

  const int g = lane >> 4;
  const _Float16* alds = As + (size_t)(lane & 15) * KDIM;
  const _Float16* brow = Bt + (size_t)(w * 16 + (lane & 15)) * KDIM;

  v8f acc = {};
#pragma unroll
  for (int k0 = 0; k0 < KDIM; k0 += 32) {
    v16h a = *(const v16h*)(alds + k0 + 16 * g);
    v16h b = *(const v16h*)(brow + k0 + 16 * g);
    acc = __builtin_amdgcn_wmma_f32_16x16x32_f16(false, a, false, b,
                                                 (short)0, acc, false, false);
  }

  const int ldc = Ntiles * 16;
  float* crow = C + (size_t)(mt * 16 + 8 * g) * ldc + w * 16 + (lane & 15);
#pragma unroll
  for (int r = 0; r < 8; ++r) crow[(size_t)r * ldc] = acc[r];
}

// ---------------- WMMA GEMM (K=H) + bias + row softmax -> assignments ----------------
__global__ void k_gemm_softmax(const _Float16* __restrict__ A, const _Float16* __restrict__ Bt,
                               const float* __restrict__ ba, float* __restrict__ S, int Mtiles) {
  int wave = (blockIdx.x * blockDim.x + threadIdx.x) >> 5;
  int lane = threadIdx.x & 31;
  if (wave >= Mtiles) return;
  int g = lane >> 4;
  const _Float16* arow = A  + (size_t)(wave * 16 + (lane & 15)) * HH;
  const _Float16* brow = Bt + (size_t)(lane & 15) * HH;

  v8f acc = {};
#pragma unroll
  for (int k0 = 0; k0 < HH; k0 += 32) {
    v16h a = *(const v16h*)(arow + k0 + 16 * g);
    v16h b = *(const v16h*)(brow + k0 + 16 * g);
    acc = __builtin_amdgcn_wmma_f32_16x16x32_f16(false, a, false, b,
                                                 (short)0, acc, false, false);
  }

  float bias = ba[lane & 15];
#pragma unroll
  for (int r = 0; r < 8; ++r) {
    float x = acc[r] + bias;
    float mx = x;
    for (int off = 8; off >= 1; off >>= 1) mx = fmaxf(mx, __shfl_xor(mx, off, 16));
    float e = __expf(x - mx);
    float sm = e;
    for (int off = 8; off >= 1; off >>= 1) sm += __shfl_xor(sm, off, 16);
    S[(size_t)(wave * 16 + r + 8 * g) * KK + (lane & 15)] = e / sm;
  }
}

// ---------------- edge aggregation: out[dst] += x[src] * dis[src]*dis[dst] ----------------
__global__ void k_agg(const float* __restrict__ x, float* __restrict__ out,
                      const int* __restrict__ src, const int* __restrict__ dst,
                      const float* __restrict__ dis) {
  int e = blockIdx.x;
  int h = threadIdx.x;
  int s = src[e], d = dst[e];
  float norm = dis[s] * dis[d];
  atomicAdd(&out[(size_t)d * HH + h], x[(size_t)s * HH + h] * norm);
}

// ---------------- combine: outf = selu(agg + dis^2*xlin + bias) (+ skip); outh = f16(outf) ----
__global__ void k_combine(const float* __restrict__ agg, const float* __restrict__ xlin,
                          const float* __restrict__ skip, const float* __restrict__ dis,
                          const float* __restrict__ bias, float* __restrict__ outf,
                          _Float16* __restrict__ outh, int has_skip) {
  size_t i = (size_t)blockIdx.x * blockDim.x + threadIdx.x;
  if (i >= (size_t)NN * HH) return;
  int n = (int)(i / HH), h = (int)(i % HH);
  float di = dis[n];
  float v = agg[i] + di * di * xlin[i] + bias[h];
  float r = selu_f(v);
  if (has_skip) r += skip[i];
  outf[i] = r;
  outh[i] = (_Float16)r;
}

// ---------------- per-node stats: csize, dS, entropy-sum, m-sum ----------------
__global__ void k_stats(const float* __restrict__ S, const float* __restrict__ degS,
                        float* __restrict__ scal) {
  __shared__ float s_cs[KK], s_ds[KK], s_ent, s_m;
  int tid = threadIdx.x;
  if (tid < KK) { s_cs[tid] = 0.f; s_ds[tid] = 0.f; }
  if (tid == 0) { s_ent = 0.f; s_m = 0.f; }
  __syncthreads();
  int n = blockIdx.x * blockDim.x + tid;
  if (n < NN) {
    float deg = degS[n];
    const float* row = S + (size_t)n * KK;
    float ent = 0.f;
#pragma unroll
    for (int k = 0; k < KK; ++k) {
      float s = row[k];
      atomicAdd(&s_cs[k], s);
      atomicAdd(&s_ds[k], deg * s);
      ent += s * logf(s + 1e-8f);
    }
    atomicAdd(&s_ent, ent);
    atomicAdd(&s_m, deg);
  }
  __syncthreads();
  if (tid < KK) {
    atomicAdd(&scal[tid], s_cs[tid]);
    atomicAdd(&scal[KK + tid], s_ds[tid]);
  }
  if (tid == 0) { atomicAdd(&scal[32], s_ent); atomicAdd(&scal[33], s_m); }
}

// ---------------- trace(S^T A S) = sum_e dot(S[src], S[dst]) ----------------
__global__ void k_trace(const float* __restrict__ S, const int* __restrict__ src,
                        const int* __restrict__ dst, float* __restrict__ scal) {
  int e = blockIdx.x * blockDim.x + threadIdx.x;
  float v = 0.f;
  if (e < EE) {
    const float* a = S + (size_t)src[e] * KK;
    const float* b = S + (size_t)dst[e] * KK;
#pragma unroll
    for (int k = 0; k < KK; ++k) v += a[k] * b[k];
  }
  for (int off = 16; off >= 1; off >>= 1) v += __shfl_xor(v, off, 32);
  if ((threadIdx.x & 31) == 0) atomicAdd(&scal[34], v);
}

// ---------------- pooled[k,d] += sum_{n in chunk} S[n,k]/(csize[k]+eps) * emb[n,d] ----------
constexpr int NCHUNK = 64;
constexpr int CHUNK  = (NN + NCHUNK - 1) / NCHUNK;
__global__ void k_pooled(const float* __restrict__ S, const float* __restrict__ emb,
                         const float* __restrict__ scal, float* __restrict__ out) {
  int tid = threadIdx.x;
  int k = tid >> 4;
  int d = (blockIdx.x << 4) + (tid & 15);
  int n0 = blockIdx.y * CHUNK;
  int n1 = n0 + CHUNK < NN ? n0 + CHUNK : NN;
  float acc = 0.f;
  for (int n = n0; n < n1; ++n)
    acc += S[(size_t)n * KK + k] * emb[(size_t)n * DD + d];
  atomicAdd(&out[(size_t)k * DD + d], acc / (scal[k] + 1e-8f));
}

// ---------------- final scalars ----------------
__global__ void k_finalize(const float* __restrict__ scal, float* __restrict__ out) {
  if (threadIdx.x != 0 || blockIdx.x != 0) return;
  float m = scal[33] * 0.5f;
  float ds2 = 0.f, cs2 = 0.f;
#pragma unroll
  for (int k = 0; k < KK; ++k) {
    ds2 += scal[KK + k] * scal[KK + k];
    cs2 += scal[k] * scal[k];
  }
  float normalizer = ds2 / (2.f * m);
  float spectral = -(scal[34] - (float)KK * normalizer) / (2.f * m);
  float collapse = sqrtf(cs2) / (float)NN * 4.f - 1.f;  // sqrt(K)=4, COLLAPSE_REG=1
  float entropy = -scal[32] / (float)NN;
  float entropy_loss = -0.1f * entropy;
  out[0] = spectral;
  out[1] = collapse;
  out[2] = spectral + collapse + entropy_loss;
  out[3] = entropy_loss;
}

// ---------------- host launcher ----------------
static inline int cdiv(long long a, long long b) { return (int)((a + b - 1) / b); }

extern "C" void kernel_launch(void* const* d_in, const int* in_sizes, int n_in,
                              void* d_out, int out_size, void* d_ws, size_t ws_size,
                              hipStream_t stream) {
  (void)in_sizes; (void)n_in; (void)out_size; (void)ws_size;
  const float* emb  = (const float*)d_in[0];
  const int*   esrc = (const int*)d_in[1];
  const int*   edst = (const int*)d_in[2];
  const float* W1 = (const float*)d_in[3];
  const float* b1 = (const float*)d_in[4];
  const float* W2 = (const float*)d_in[5];
  const float* b2 = (const float*)d_in[6];
  const float* Wa = (const float*)d_in[7];
  const float* ba = (const float*)d_in[8];

  float* wsf = (float*)d_ws;
  _Float16* wsh = (_Float16*)(wsf + F32_END);

  float* scal = wsf + SCAL_O;
  float* degd = wsf + DEGD_O;
  float* degs = wsf + DEGS_O;
  float* dis  = wsf + DIS_O;
  float* B1   = wsf + B1_O;
  float* B2   = wsf + B2_O;
  float* B3   = wsf + B3_O;
  _Float16* Xh  = wsh + XH_O;
  _Float16* Hh  = wsh + HF_O;
  _Float16* W1t = wsh + W1T_O;
  _Float16* W2t = wsh + W2T_O;
  _Float16* WaT = wsh + WAT_O;

  float* S      = (float*)d_out;                 // [N,K]
  float* pooled = S + (size_t)NN * KK;           // [K,D]
  float* sc_out = pooled + (size_t)KK * DD;      // 4 scalars

  // 1. zero scratch regions (scalars + degrees; both agg accumulators; pooled output)
  k_zero<<<cdiv(DIS_O, 256), 256, 0, stream>>>(wsf, DIS_O);
  k_zero<<<cdiv((size_t)2 * NN * HH, 256), 256, 0, stream>>>(B2, (size_t)2 * NN * HH);
  k_zero<<<cdiv(KK * DD, 256), 256, 0, stream>>>(pooled, (size_t)KK * DD);

  // 2. degrees + symmetric norm
  k_deg<<<cdiv(EE, 256), 256, 0, stream>>>(esrc, edst, degd, degs);
  k_dis<<<cdiv(NN, 256), 256, 0, stream>>>(degd, dis);

  // 3. f16 conversions / weight transposes
  k_cvt_f16<<<cdiv((size_t)NN * DD, 256), 256, 0, stream>>>(emb, Xh, (size_t)NN * DD);
  k_transpose<<<cdiv(DD * HH, 256), 256, 0, stream>>>(W1, W1t, DD, HH);
  k_transpose<<<cdiv(HH * HH, 256), 256, 0, stream>>>(W2, W2t, HH, HH);
  k_transpose<<<cdiv(HH * KK, 256), 256, 0, stream>>>(Wa, WaT, HH, KK);

  // 4. layer 1: X @ W1 -> B1 ; aggregate -> B2 ; combine+selu -> B1 (skip), Hh
  k_gemm_wmma_lds<DD><<<NN / 16, 512, 0, stream>>>(Xh, W1t, B1, HH / 16);
  k_agg<<<EE, HH, 0, stream>>>(B1, B2, esrc, edst, dis);
  k_combine<<<cdiv((size_t)NN * HH, 256), 256, 0, stream>>>(B2, B1, nullptr, dis, b1,
                                                            B1, Hh, 0);

  // 5. layer 2: Hh @ W2 -> B2 ; aggregate -> B3 ; combine+selu+skip -> B2, Hh
  k_gemm_wmma_lds<HH><<<NN / 16, 512, 0, stream>>>(Hh, W2t, B2, HH / 16);
  k_agg<<<EE, HH, 0, stream>>>(B2, B3, esrc, edst, dis);
  k_combine<<<cdiv((size_t)NN * HH, 256), 256, 0, stream>>>(B3, B2, B1, dis, b2,
                                                            B2, Hh, 1);

  // 6. assignment head: softmax(Hh @ Wa + ba) -> S (d_out)
  k_gemm_softmax<<<cdiv(NN / 16, 8), 256, 0, stream>>>(Hh, WaT, ba, S, NN / 16);

  // 7. statistics + losses + pooled
  k_stats<<<cdiv(NN, 256), 256, 0, stream>>>(S, degs, scal);
  k_trace<<<cdiv(EE, 256), 256, 0, stream>>>(S, esrc, edst, scal);
  {
    dim3 pg(DD / 16, NCHUNK);
    k_pooled<<<pg, 256, 0, stream>>>(S, emb, scal, pooled);
  }
  k_finalize<<<1, 1, 0, stream>>>(scal, sc_out);
}
